// KanvolutionLayer_44581760532956
// MI455X (gfx1250) — compile-verified, matrix operand order checked by reading, not run on hardware
//
#include <hip/hip_runtime.h>
#include <stdint.h>

// ---------------- types ----------------
typedef __attribute__((ext_vector_type(16))) __bf16 bf16x16;
typedef __attribute__((ext_vector_type(8)))  float  f32x8;

union BFrag {
    uint4   q[2];
    bf16x16 v;
};
static_assert(sizeof(BFrag) == 32, "frag is 32B");

// ---------------- problem constants ----------------
#define C_IN    256
#define C_OUT   256
#define P_ORD   5
#define Q_ORD   4
#define HW      3136        // 56*56
#define BATCH   32
#define NT      64          // spatial tile width (columns)
#define TILES_PER_IMG (HW / NT)          // 49
#define NTILES  (BATCH * TILES_PER_IMG)  // 1568
#define KCH     40          // max K-chunks of 32 (P region: 5*256/32)
#define KCQ     32          // Q region K-chunks (4*256/32)

__device__ __forceinline__ void split_bf16(float w, unsigned short& hi, unsigned short& lo) {
    __bf16 h = (__bf16)w;
    hi = __builtin_bit_cast(unsigned short, h);
    __bf16 l = (__bf16)(w - (float)h);
    lo = __builtin_bit_cast(unsigned short, l);
}

// ============================================================
// Pre-pass: split fp32 weights into hi/lo bf16, stored in the
// exact WMMA A-fragment lane layout (ISA 7.12.2, 16-bit A 16x32):
//   lane L (h = L>>4, m = mt*16 + (L&15)) holds 16 bf16:
//   dwords 0..3 -> k = h*8 + {0..7},  dwords 4..7 -> k = 16 + h*8 + {0..7}
// Fragment (mt, kc) at uint4 index ((mt*KCH + kc)*32 + lane)*2.
// Row index m: 0..255 => Wp rows, 256..511 => Wq rows.
// K index: k_global = kc*32 + k_local, with p = kc/8, c = (kc%8)*32 + k_local.
// ============================================================
__global__ __launch_bounds__(256) void kan_prepack(const float* __restrict__ Wp,
                                                   const float* __restrict__ Wq,
                                                   uint4* __restrict__ aHi,
                                                   uint4* __restrict__ aLo) {
    int g    = blockIdx.x * 256 + threadIdx.x;   // one lane-fragment per thread
    int lane = g & 31;
    int t2   = g >> 5;
    int kc   = t2 % KCH;
    int mt   = t2 / KCH;
    if (mt >= 32) return;

    int m = mt * 16 + (lane & 15);
    int h = lane >> 4;

    bool qRegion = (m >= C_OUT);
    bool valid   = !(qRegion && kc >= KCQ);      // Q region only has 32 k-chunks
    int  p       = kc >> 3;
    int  cbase   = (kc & 7) * 32;
    const float* W = qRegion
        ? (Wq + (size_t)p * C_OUT * C_IN + (size_t)(m - C_OUT) * C_IN)
        : (Wp + (size_t)p * C_OUT * C_IN + (size_t)m * C_IN);

    unsigned int hiw[8], low[8];
    #pragma unroll
    for (int d = 0; d < 8; ++d) {
        int k0  = ((d & 4) ? 16 : 0) + h * 8 + (d & 3) * 2;   // even k of the pair
        float w0 = valid ? W[cbase + k0]     : 0.0f;
        float w1 = valid ? W[cbase + k0 + 1] : 0.0f;
        unsigned short h0, l0, h1, l1;
        split_bf16(w0, h0, l0);
        split_bf16(w1, h1, l1);
        hiw[d] = ((unsigned int)h1 << 16) | h0;
        low[d] = ((unsigned int)l1 << 16) | l0;
    }

    size_t fi = ((size_t)(mt * KCH + kc) * 32 + lane) * 2;
    uint4 u;
    u.x = hiw[0]; u.y = hiw[1]; u.z = hiw[2]; u.w = hiw[3];  aHi[fi]     = u;
    u.x = hiw[4]; u.y = hiw[5]; u.z = hiw[6]; u.w = hiw[7];  aHi[fi + 1] = u;
    u.x = low[0]; u.y = low[1]; u.z = low[2]; u.w = low[3];  aLo[fi]     = u;
    u.x = low[4]; u.y = low[5]; u.z = low[6]; u.w = low[7];  aLo[fi + 1] = u;
}

// ============================================================
// Main fused kernel: one block = 64 spatial columns x all 512 rows.
// 512 threads = 16 wave32: waves 0..7 -> P rows (o 0..255),
// waves 8..15 -> Q rows. Each wave: 2 row-tiles x 4 col-tiles,
// 3 split-product WMMAs per (tile, k-chunk).
// LDS (64KB): B-fragments hi (32KB) + lo (32KB); aliased afterwards
// as the 256x64 fp32 sum_q buffer for the fusion epilogue.
// ============================================================
__global__ __launch_bounds__(512) void kan_main(const float* __restrict__ x,
                                                const uint4* __restrict__ aHi,
                                                const uint4* __restrict__ aLo,
                                                const float* __restrict__ bp,
                                                float* __restrict__ out) {
    __shared__ __align__(16) unsigned int smem[16384];   // 64 KB
    unsigned int* sBhi = smem;            // 8192 dwords
    unsigned int* sBlo = smem + 8192;     // 8192 dwords

    const int tid  = threadIdx.x;
    const int lane = tid & 31;
    const int wv   = tid >> 5;

    const int tile = blockIdx.x;
    const int b    = tile / TILES_PER_IMG;
    const int hw0  = (tile % TILES_PER_IMG) * NT;
    const float* xb = x + (size_t)b * C_IN * HW + hw0;

    // producer ownership: n column + 16 channel-pairs (c strided for coalescing)
    const int pn  = tid & 63;     // column within tile
    const int cpg = tid >> 6;     // 0..7 channel-pair group

    // consumer assignment
    const bool isQ = (wv >= 8);
    const int  rt0 = isQ ? 16 + (wv - 8) * 2 : wv * 2;

    f32x8 acc[2][4];
    const f32x8 zero = {0.f, 0.f, 0.f, 0.f, 0.f, 0.f, 0.f, 0.f};
    #pragma unroll
    for (int i = 0; i < 2; ++i)
        #pragma unroll
        for (int j = 0; j < 4; ++j)
            acc[i][j] = zero;

    for (int pp = 0; pp < P_ORD; ++pp) {
        // ---- produce x^(pp+1) hi/lo B-fragments into LDS ----
        // B layout (16-bit B 32x16, mirror of A): lane = ((k>>3)&1)*16 + n%16,
        // dword = (k&16 ? 4 : 0) + (k&7)/2, low half = even k.
        #pragma unroll 4
        for (int i = 0; i < 16; ++i) {
            int cp = cpg * 16 + i;
            int c0 = cp * 2;
            float x0 = xb[(size_t)c0 * HW + pn];
            float x1 = xb[(size_t)(c0 + 1) * HW + pn];
            float v0 = x0, v1 = x1;
            for (int j = 0; j < pp; ++j) { v0 *= x0; v1 *= x1; }
            unsigned short h0, l0, h1, l1;
            split_bf16(v0, h0, l0);
            split_bf16(v1, h1, l1);
            int kl    = c0 & 31;
            int cblk  = c0 >> 5;
            int laneb = ((kl >> 3) & 1) * 16 + (pn & 15);
            int nt4   = pn >> 4;
            int dw    = ((kl & 16) ? 4 : 0) + ((kl & 7) >> 1);
            int uidx  = ((cblk * 4 + nt4) * 32 + laneb) * 8 + dw;
            sBhi[uidx] = ((unsigned int)h1 << 16) | h0;
            sBlo[uidx] = ((unsigned int)l1 << 16) | l0;
        }
        __syncthreads();

        // ---- consume: GEMM over this power's 8 k-chunks ----
        if (!isQ || pp < Q_ORD) {
            const uint4* sB4hi = (const uint4*)sBhi;
            const uint4* sB4lo = (const uint4*)sBlo;
            for (int cblk = 0; cblk < 8; ++cblk) {
                const int kc = pp * 8 + cblk;
                BFrag a0h, a0l, a1h, a1l;
                size_t f0 = ((size_t)(rt0 * KCH + kc) * 32 + lane) * 2;
                size_t f1 = ((size_t)((rt0 + 1) * KCH + kc) * 32 + lane) * 2;
                a0h.q[0] = aHi[f0];     a0h.q[1] = aHi[f0 + 1];
                a0l.q[0] = aLo[f0];     a0l.q[1] = aLo[f0 + 1];
                a1h.q[0] = aHi[f1];     a1h.q[1] = aHi[f1 + 1];
                a1l.q[0] = aLo[f1];     a1l.q[1] = aLo[f1 + 1];
                #pragma unroll
                for (int ct = 0; ct < 4; ++ct) {
                    int bi = ((cblk * 4 + ct) * 32 + lane) * 2;
                    BFrag bh, bl;
                    bh.q[0] = sB4hi[bi]; bh.q[1] = sB4hi[bi + 1];
                    bl.q[0] = sB4lo[bi]; bl.q[1] = sB4lo[bi + 1];
                    // split-precision: hi*hi + hi*lo + lo*hi, fp32 accumulate
                    acc[0][ct] = __builtin_amdgcn_wmma_f32_16x16x32_bf16(
                        false, a0h.v, false, bh.v, (short)0, acc[0][ct], false, false);
                    acc[0][ct] = __builtin_amdgcn_wmma_f32_16x16x32_bf16(
                        false, a0h.v, false, bl.v, (short)0, acc[0][ct], false, false);
                    acc[0][ct] = __builtin_amdgcn_wmma_f32_16x16x32_bf16(
                        false, a0l.v, false, bh.v, (short)0, acc[0][ct], false, false);
                    acc[1][ct] = __builtin_amdgcn_wmma_f32_16x16x32_bf16(
                        false, a1h.v, false, bh.v, (short)0, acc[1][ct], false, false);
                    acc[1][ct] = __builtin_amdgcn_wmma_f32_16x16x32_bf16(
                        false, a1h.v, false, bl.v, (short)0, acc[1][ct], false, false);
                    acc[1][ct] = __builtin_amdgcn_wmma_f32_16x16x32_bf16(
                        false, a1l.v, false, bh.v, (short)0, acc[1][ct], false, false);
                }
            }
        }
        __syncthreads();
    }

    // ---- epilogue: Q waves deposit sum_q into LDS (aliases B buffers) ----
    float* qbuf = (float*)smem;                 // 256 x 64 fp32 = 64 KB
    const int hb = lane >> 4;
    const int nn = lane & 15;
    if (isQ) {
        #pragma unroll
        for (int r2 = 0; r2 < 2; ++r2) {
            int ob = (rt0 - 16 + r2) * 16;
            #pragma unroll
            for (int ct = 0; ct < 4; ++ct)
                #pragma unroll
                for (int r = 0; r < 8; ++r) {
                    int o = ob + r + 8 * hb;    // C/D layout: M = r + 8*(lane>>4)
                    qbuf[o * NT + ct * 16 + nn] = acc[r2][ct][r];
                }
        }
    }
    __syncthreads();

    // ---- P waves fuse bias + 1/(1+|q|) and write out ----
    if (!isQ) {
        float* ob_ = out + (size_t)b * C_OUT * HW + hw0;
        #pragma unroll
        for (int r2 = 0; r2 < 2; ++r2) {
            int obase = (rt0 + r2) * 16;
            #pragma unroll
            for (int ct = 0; ct < 4; ++ct)
                #pragma unroll
                for (int r = 0; r < 8; ++r) {
                    int o  = obase + r + 8 * hb;
                    int nl = ct * 16 + nn;
                    float q = qbuf[o * NT + nl];
                    float v = (acc[r2][ct][r] + bp[o]) / (1.0f + fabsf(q));
                    ob_[(size_t)o * HW + nl] = v;
                }
        }
    }
}

// ============================================================
extern "C" void kernel_launch(void* const* d_in, const int* in_sizes, int n_in,
                              void* d_out, int out_size, void* d_ws, size_t ws_size,
                              hipStream_t stream) {
    (void)in_sizes; (void)n_in; (void)out_size; (void)ws_size;
    const float* x  = (const float*)d_in[0];
    const float* Wp = (const float*)d_in[1];
    const float* bp = (const float*)d_in[2];
    const float* Wq = (const float*)d_in[3];
    float* out = (float*)d_out;

    // workspace: pre-swizzled A fragments, hi then lo
    const size_t fragBytes = (size_t)32 * KCH * 32 * 32;   // 1.25 MB each
    uint4* aHi = (uint4*)d_ws;
    uint4* aLo = (uint4*)((char*)d_ws + fragBytes);

    kan_prepack<<<(32 * KCH * 32) / 256, 256, 0, stream>>>(Wp, Wq, aHi, aLo);
    kan_main<<<NTILES, 512, 0, stream>>>(x, aHi, aLo, bp, out);
}
